// Mamba_block_67920612819445
// MI455X (gfx1250) — matrix operand order repaired and checked by reading, not compiled, not run on hardware
//
#include <hip/hip_runtime.h>
#include <hip/hip_bf16.h>

#define DIMC   64
#define DINNER 128
#define DSTATE 16
#define DTRANK 4
#define LSEQ   9216   // 96*96
#define NBATCH 4
#define HWD    96
#define NCHUNK 96     // parallel-scan chunks per sequence
#define CLEN   96     // NCHUNK * CLEN == LSEQ

typedef __attribute__((ext_vector_type(16))) _Float16 v16h;
typedef __attribute__((ext_vector_type(8)))  float    v8f;

// ---------------- WMMA fragment helpers (ISA 7.12.2 layouts, wave32) -------

// A: 16x32 f16, row-major X (M x ldk). lane<16 -> K 0..7,16..23 ; lane>=16 -> K 8..15,24..31
__device__ __forceinline__ v16h frag_a(const _Float16* __restrict__ X, int ldk,
                                       int row0, int k0, int lane) {
  int m  = lane & 15;
  int kb = (lane < 16) ? 0 : 8;
  const _Float16* p = X + (size_t)(row0 + m) * ldk + k0;
  v16h a;
#pragma unroll
  for (int i = 0; i < 8; ++i) a[i] = p[kb + i];
#pragma unroll
  for (int i = 0; i < 8; ++i) a[8 + i] = p[16 + kb + i];
  return a;
}

// B: 32x16 f16 where b(k,n) = W[n][k], W row-major (N x ldk).
// lane<16 -> K 0..7,16..23 ; lane>=16 -> K 8..15,24..31 ; col = lane%16
__device__ __forceinline__ v16h frag_b(const _Float16* __restrict__ W, int ldk,
                                       int col0, int k0, int lane) {
  int n  = lane & 15;
  int kb = (lane < 16) ? 0 : 8;
  const _Float16* p = W + (size_t)(col0 + n) * ldk + k0;
  v16h b;
#pragma unroll
  for (int i = 0; i < 8; ++i) b[i] = p[kb + i];
#pragma unroll
  for (int i = 0; i < 8; ++i) b[8 + i] = p[16 + kb + i];
  return b;
}

// ---------------- K0: weight prep (f32 -> f16, A = -exp(A_log)) ------------

__global__ void prep_kernel(const float* __restrict__ inW, const float* __restrict__ xpW,
                            const float* __restrict__ outW, const float* __restrict__ A_log,
                            _Float16* __restrict__ inW16, _Float16* __restrict__ xpW16,
                            _Float16* __restrict__ outW16, float* __restrict__ Aneg) {
  int t = blockIdx.x * blockDim.x + threadIdx.x;
  int stride = blockDim.x * gridDim.x;
  for (int i = t; i < 256 * 64; i += stride) inW16[i] = (_Float16)inW[i];
  for (int i = t; i < 48 * 128; i += stride) {
    int n = i >> 7, k = i & 127;
    xpW16[i] = (n < 36) ? (_Float16)xpW[n * 128 + k] : (_Float16)0.f;
  }
  for (int i = t; i < 64 * 128; i += stride) outW16[i] = (_Float16)outW[i];
  for (int i = t; i < 128 * 16; i += stride) Aneg[i] = -__expf(A_log[i]);
}

// ---------------- K1: reflect-pad conv2d (k=4,s=2) + bias + LayerNorm ------

__global__ __launch_bounds__(64) void conv_ln_kernel(
    const float* __restrict__ x, const float* __restrict__ cw,
    const float* __restrict__ cb, const float* __restrict__ g,
    const float* __restrict__ bta, float* __restrict__ xc,
    _Float16* __restrict__ nrm16) {
  __shared__ float patch[512];
  __shared__ float red[64];
  __shared__ float red2[64];
  int p = blockIdx.x;
  int b = p / LSEQ;
  int l = p - b * LSEQ;
  int oh = l / HWD, ow = l - oh * HWD;
  int tid = threadIdx.x;
#pragma unroll
  for (int k = 0; k < 8; ++k) {
    int idx = tid + k * 64;
    int cin = idx >> 4;
    int r = idx & 15;
    int kh = r >> 2, kw = r & 3;
    int ih = oh * 2 - 1 + kh;
    int iw = ow * 2 - 1 + kw;
    ih = (ih < 0) ? -ih : ((ih > 191) ? 382 - ih : ih);
    iw = (iw < 0) ? -iw : ((iw > 191) ? 382 - iw : iw);
    patch[idx] = x[(((size_t)b * 32 + cin) * 192 + ih) * 192 + iw];
  }
  __syncthreads();
  float acc = cb[tid];
  const float* wr = cw + (size_t)tid * 512;
#pragma unroll 8
  for (int i = 0; i < 512; ++i) acc = fmaf(wr[i], patch[i], acc);
  size_t xci = ((size_t)b * DIMC + tid) * LSEQ + l;
  xc[xci] = acc;
  red[tid] = acc;
  red2[tid] = acc * acc;
  __syncthreads();
  for (int s = 32; s > 0; s >>= 1) {
    if (tid < s) { red[tid] += red[tid + s]; red2[tid] += red2[tid + s]; }
    __syncthreads();
  }
  float mu = red[0] * (1.f / 64.f);
  float var = red2[0] * (1.f / 64.f) - mu * mu;
  float nv = (acc - mu) * rsqrtf(var + 1e-5f) * g[tid] + bta[tid];
  nrm16[(size_t)p * DIMC + tid] = (_Float16)nv;
}

// ---------------- K2: in_proj GEMM  (M=36864, N=256, K=64) via WMMA --------

__global__ __launch_bounds__(256) void inproj_kernel(const _Float16* __restrict__ A,
                                                     const _Float16* __restrict__ W,
                                                     float* __restrict__ xz) {
  int lane = threadIdx.x & 31;
  int wv = threadIdx.x >> 5;
  int row0 = blockIdx.x * 16;
  int col0 = (blockIdx.y * 8 + wv) * 16;
  v8f c = {};
#pragma unroll
  for (int k0 = 0; k0 < 64; k0 += 32) {
    v16h af = frag_a(A, 64, row0, k0, lane);
    v16h bf = frag_b(W, 64, col0, k0, lane);
    c = __builtin_amdgcn_wmma_f32_16x16x32_f16(false, af, false, bf, (short)0, c, false, false);
  }
  int n = col0 + (lane & 15);
  int mb = (lane < 16) ? 0 : 8;
#pragma unroll
  for (int j = 0; j < 8; ++j) xz[(size_t)(row0 + mb + j) * 256 + n] = c[j];
}

// ---------------- K3: causal depthwise conv1d (k=4) + SiLU -----------------

__global__ void conv1d_kernel(const float* __restrict__ xz, const float* __restrict__ w1,
                              const float* __restrict__ b1, float* __restrict__ xm,
                              _Float16* __restrict__ xm16) {
  int i = blockIdx.x * blockDim.x + threadIdx.x;
  if (i >= NBATCH * LSEQ * DINNER) return;
  int d = i & 127;
  int bl = i >> 7;
  int b = bl / LSEQ;
  int l = bl - b * LSEQ;
  float acc = b1[d];
#pragma unroll
  for (int j = 0; j < 4; ++j) {
    int t = l - 3 + j;
    if (t >= 0) acc = fmaf(w1[d * 4 + j], xz[((size_t)(b * LSEQ + t)) * 256 + d], acc);
  }
  float s = acc / (1.f + __expf(-acc));   // SiLU
  xm[i] = s;
  xm16[i] = (_Float16)s;
}

// ---------------- K4: x_proj GEMM (M=36864, N=48 padded, K=128) ------------

__global__ __launch_bounds__(96) void xproj_kernel(const _Float16* __restrict__ A,
                                                   const _Float16* __restrict__ W,
                                                   float* __restrict__ xdbl) {
  int lane = threadIdx.x & 31;
  int wv = threadIdx.x >> 5;
  int row0 = blockIdx.x * 16;
  int col0 = wv * 16;
  v8f c = {};
#pragma unroll
  for (int k0 = 0; k0 < 128; k0 += 32) {
    v16h af = frag_a(A, 128, row0, k0, lane);
    v16h bf = frag_b(W, 128, col0, k0, lane);
    c = __builtin_amdgcn_wmma_f32_16x16x32_f16(false, af, false, bf, (short)0, c, false, false);
  }
  int n = col0 + (lane & 15);
  int mb = (lane < 16) ? 0 : 8;
#pragma unroll
  for (int j = 0; j < 8; ++j) xdbl[(size_t)(row0 + mb + j) * 48 + n] = c[j];
}

// ---------------- chunked parallel scan ------------------------------------
// Recurrence h_t = dA_t * h_{t-1} + dBx_t is associative. Split L into
// NCHUNK chunks of CLEN. Pass1: per-chunk composed transition (P = prod dA,
// S = zero-init local scan endpoint). Pass2: tiny serial combine over chunk
// summaries to get each chunk's true initial state. Pass3: replay chunks
// from their initial state and emit the gated output. Serial depth drops
// from 9216 to ~96+96+96; wave parallelism rises from 16 to 1536 waves.

// Pass 1: grid = NBATCH*NCHUNK blocks x 128 threads (one d per thread).
__global__ __launch_bounds__(128) void scan_pass1_kernel(
    const float* __restrict__ xdbl, const float* __restrict__ xm,
    const float* __restrict__ Aneg, const float* __restrict__ dtw,
    const float* __restrict__ dtb, float* __restrict__ chunkP,
    float* __restrict__ chunkS) {
  int blk = blockIdx.x;
  int b = blk / NCHUNK;
  int ck = blk - b * NCHUNK;
  int d = threadIdx.x;
  float a[16], P[16], h[16];
#pragma unroll
  for (int s = 0; s < 16; ++s) { a[s] = Aneg[d * 16 + s]; P[s] = 1.f; h[s] = 0.f; }
  float w0 = dtw[d * 4 + 0], w1 = dtw[d * 4 + 1];
  float w2 = dtw[d * 4 + 2], w3 = dtw[d * 4 + 3];
  float bb = dtb[d];
  int t0 = ck * CLEN;
  const float* drow = xdbl + ((size_t)b * LSEQ + t0) * 48;
  const float* xmrow = xm + ((size_t)b * LSEQ + t0) * 128 + d;
  for (int t = 0; t < CLEN; ++t) {
    const float* r = drow + (size_t)t * 48;
    __builtin_prefetch(r + 96, 0, 0);            // global_prefetch_b8
    float u = fmaf(w0, r[0], fmaf(w1, r[1], fmaf(w2, r[2], fmaf(w3, r[3], bb))));
    float dt = (u > 20.f) ? u : __logf(1.f + __expf(u));
    float v = xmrow[(size_t)t * 128];
    float dtx = dt * v;
#pragma unroll
    for (int s = 0; s < 16; ++s) {
      float e = __expf(dt * a[s]);
      P[s] *= e;
      h[s] = fmaf(e, h[s], dtx * r[4 + s]);
    }
  }
  size_t base = ((size_t)(b * NCHUNK + ck) * 128 + d) * 16;
#pragma unroll
  for (int s = 0; s < 16; ++s) { chunkP[base + s] = P[s]; chunkS[base + s] = h[s]; }
}

// Pass 2: 512 chains, one lane each; serial over NCHUNK chunk summaries.
__global__ __launch_bounds__(32) void scan_pass2_kernel(
    const float* __restrict__ chunkP, const float* __restrict__ chunkS,
    float* __restrict__ hinit) {
  int w = blockIdx.x;               // 0..15
  int b = w >> 2;
  int d = ((w & 3) << 5) + threadIdx.x;
  float h[16];
#pragma unroll
  for (int s = 0; s < 16; ++s) h[s] = 0.f;
  for (int ck = 0; ck < NCHUNK; ++ck) {
    size_t base = ((size_t)(b * NCHUNK + ck) * 128 + d) * 16;
#pragma unroll
    for (int s = 0; s < 16; ++s) {
      hinit[base + s] = h[s];
      h[s] = fmaf(chunkP[base + s], h[s], chunkS[base + s]);
    }
  }
}

// Pass 3: replay chunks from true initial state, emit gated f16 operand.
__global__ __launch_bounds__(128) void scan_pass3_kernel(
    const float* __restrict__ xdbl, const float* __restrict__ xm,
    const float* __restrict__ xz, const float* __restrict__ Aneg,
    const float* __restrict__ dtw, const float* __restrict__ dtb,
    const float* __restrict__ Dp, const float* __restrict__ hinit,
    _Float16* __restrict__ yg16) {
  int blk = blockIdx.x;
  int b = blk / NCHUNK;
  int ck = blk - b * NCHUNK;
  int d = threadIdx.x;
  float a[16], h[16];
  size_t hbase = ((size_t)(b * NCHUNK + ck) * 128 + d) * 16;
#pragma unroll
  for (int s = 0; s < 16; ++s) { a[s] = Aneg[d * 16 + s]; h[s] = hinit[hbase + s]; }
  float w0 = dtw[d * 4 + 0], w1 = dtw[d * 4 + 1];
  float w2 = dtw[d * 4 + 2], w3 = dtw[d * 4 + 3];
  float bb = dtb[d];
  float Dd = Dp[d];
  int t0 = ck * CLEN;
  const float* drow = xdbl + ((size_t)b * LSEQ + t0) * 48;
  const float* xmrow = xm + ((size_t)b * LSEQ + t0) * 128 + d;
  const float* zrow = xz + ((size_t)b * LSEQ + t0) * 256 + 128 + d;
  _Float16* yrow = yg16 + ((size_t)b * LSEQ + t0) * 128 + d;
  for (int t = 0; t < CLEN; ++t) {
    const float* r = drow + (size_t)t * 48;
    __builtin_prefetch(r + 96, 0, 0);            // global_prefetch_b8
    float u = fmaf(w0, r[0], fmaf(w1, r[1], fmaf(w2, r[2], fmaf(w3, r[3], bb))));
    float dt = (u > 20.f) ? u : __logf(1.f + __expf(u));
    float v = xmrow[(size_t)t * 128];
    float dtx = dt * v;
    float y = 0.f;
#pragma unroll
    for (int s = 0; s < 16; ++s) {
      float e = __expf(dt * a[s]);
      float hs = fmaf(e, h[s], dtx * r[4 + s]);
      h[s] = hs;
      y = fmaf(hs, r[20 + s], y);
    }
    float z = zrow[(size_t)t * 256];
    float gz = z / (1.f + __expf(-z));
    yrow[(size_t)t * 128] = (_Float16)((y + v * Dd) * gz);
  }
}

// ---------------- K7: out_proj GEMM + fused residual (+2*xc) ---------------

__global__ __launch_bounds__(128) void outproj_kernel(const _Float16* __restrict__ A,
                                                      const _Float16* __restrict__ W,
                                                      const float* __restrict__ xc,
                                                      float* __restrict__ out) {
  int lane = threadIdx.x & 31;
  int wv = threadIdx.x >> 5;
  int row0 = blockIdx.x * 16;
  int col0 = wv * 16;
  v8f c = {};
#pragma unroll
  for (int k0 = 0; k0 < 128; k0 += 32) {
    v16h af = frag_a(A, 128, row0, k0, lane);
    v16h bf = frag_b(W, 128, col0, k0, lane);
    c = __builtin_amdgcn_wmma_f32_16x16x32_f16(false, af, false, bf, (short)0, c, false, false);
  }
  int n = col0 + (lane & 15);
  int mb = (lane < 16) ? 0 : 8;
#pragma unroll
  for (int j = 0; j < 8; ++j) {
    int row = row0 + mb + j;
    int b = row / LSEQ;
    int l = row - b * LSEQ;
    size_t oi = ((size_t)b * DIMC + n) * LSEQ + l;
    out[oi] = c[j] + 2.f * xc[oi];   // seq residual == xc transposed, + outer xc
  }
}

// ---------------- host launcher --------------------------------------------

extern "C" void kernel_launch(void* const* d_in, const int* in_sizes, int n_in,
                              void* d_out, int out_size, void* d_ws, size_t ws_size,
                              hipStream_t stream) {
  const float* x      = (const float*)d_in[0];
  const float* conv_w = (const float*)d_in[1];
  const float* conv_b = (const float*)d_in[2];
  const float* ln_g   = (const float*)d_in[3];
  const float* ln_b   = (const float*)d_in[4];
  const float* inW    = (const float*)d_in[5];
  const float* c1w    = (const float*)d_in[6];
  const float* c1b    = (const float*)d_in[7];
  const float* xpW    = (const float*)d_in[8];
  const float* dtw    = (const float*)d_in[9];
  const float* dtb    = (const float*)d_in[10];
  const float* A_log  = (const float*)d_in[11];
  const float* Dp     = (const float*)d_in[12];
  const float* outW   = (const float*)d_in[13];

  char* ws = (char*)d_ws;
  size_t off = 0;
  auto alloc = [&](size_t bytes) -> void* {
    void* p = ws + off;
    off += (bytes + 255) & ~(size_t)255;
    return p;
  };
  _Float16* inW16  = (_Float16*)alloc((size_t)256 * 64 * 2);
  _Float16* xpW16  = (_Float16*)alloc((size_t)48 * 128 * 2);
  _Float16* outW16 = (_Float16*)alloc((size_t)64 * 128 * 2);
  float*    Aneg   = (float*)alloc((size_t)128 * 16 * 4);
  float*    xc     = (float*)alloc((size_t)NBATCH * DIMC * LSEQ * 4);      // 9.4 MB
  _Float16* nrm16  = (_Float16*)alloc((size_t)NBATCH * LSEQ * DIMC * 2);   // 4.7 MB
  float*    xz     = (float*)alloc((size_t)NBATCH * LSEQ * 256 * 4);       // 37.7 MB
  float*    xm     = (float*)alloc((size_t)NBATCH * LSEQ * 128 * 4);       // 18.9 MB
  _Float16* xm16   = (_Float16*)alloc((size_t)NBATCH * LSEQ * 128 * 2);    // 9.4 MB
  float*    xdbl   = (float*)alloc((size_t)NBATCH * LSEQ * 48 * 4);        // 7.1 MB
  _Float16* yg16   = (_Float16*)alloc((size_t)NBATCH * LSEQ * 128 * 2);    // 9.4 MB
  float*    chunkP = (float*)alloc((size_t)NBATCH * NCHUNK * 128 * 16 * 4); // 3.1 MB
  float*    chunkS = (float*)alloc((size_t)NBATCH * NCHUNK * 128 * 16 * 4); // 3.1 MB
  float*    hinit  = (float*)alloc((size_t)NBATCH * NCHUNK * 128 * 16 * 4); // 3.1 MB
  float* out = (float*)d_out;

  const int M = NBATCH * LSEQ;          // 36864 rows, divisible by 16

  prep_kernel<<<64, 256, 0, stream>>>(inW, xpW, outW, A_log, inW16, xpW16, outW16, Aneg);
  conv_ln_kernel<<<M, 64, 0, stream>>>(x, conv_w, conv_b, ln_g, ln_b, xc, nrm16);
  {
    dim3 g(M / 16, 2);                  // 16 N-tiles over 8 waves x 2 blocks
    inproj_kernel<<<g, 256, 0, stream>>>(nrm16, inW16, xz);
  }
  {
    int n = M * DINNER;
    conv1d_kernel<<<(n + 255) / 256, 256, 0, stream>>>(xz, c1w, c1b, xm, xm16);
  }
  xproj_kernel<<<M / 16, 96, 0, stream>>>(xm16, xpW16, xdbl);
  scan_pass1_kernel<<<NBATCH * NCHUNK, 128, 0, stream>>>(xdbl, xm, Aneg, dtw, dtb,
                                                         chunkP, chunkS);
  scan_pass2_kernel<<<16, 32, 0, stream>>>(chunkP, chunkS, hinit);
  scan_pass3_kernel<<<NBATCH * NCHUNK, 128, 0, stream>>>(xdbl, xm, xz, Aneg, dtw, dtb,
                                                         Dp, hinit, yg16);
  outproj_kernel<<<M / 16, 128, 0, stream>>>(yg16, outW16, xc, out);
}
